// Predictor_31001073943120
// MI455X (gfx1250) — matrix operand order repaired and checked by reading, not compiled, not run on hardware
//
#include <hip/hip_runtime.h>
#include <cstdint>

// ---------------------------------------------------------------------------
// MI455X (gfx1250) implementation: f16 WMMA GEMMs with double-buffered
// TDM-staged LDS tiles (DMA for tile k+1 overlaps compute of tile k),
// fused epilogues (silu*u, ls-residual, sigmoid gate), wave32 attention core.
// Memory-bound workload (~0.93 TFLOP vs ~1.5-2 GB traffic @ 23.3 TB/s), so
// everything runs in f16 with fp32 accumulation and fused elementwise ops.
// ---------------------------------------------------------------------------

typedef __attribute__((ext_vector_type(16))) _Float16 v16h;
typedef __attribute__((ext_vector_type(8)))  _Float16 v8h;
typedef __attribute__((ext_vector_type(8)))  float    v8f;
typedef __attribute__((ext_vector_type(4)))  unsigned int v4u;
typedef __attribute__((ext_vector_type(8)))  int v8i;
typedef __attribute__((ext_vector_type(4)))  int v4i;

#define DEVINL __device__ __forceinline__

// ---- Tensor Data Mover: 2D tile (tile_w elems per row, tile_h rows) -------
// D# group0/group1 packed per CDNA5 ISA.
// data_size=1 (2-byte elements), no multicast, no padding, no iteration.
// 6-arg builtin on this toolchain:
//   (uint32x4 g0, int32x8 g1, int32x4 g2, int32x4 g3, int32x8 extra, i32 cpol)
DEVINL void tdm_load_2d(unsigned lds_addr, const void* gptr,
                        unsigned tile_w, unsigned tile_h,
                        unsigned long long row_stride_elems) {
  unsigned long long ga = (unsigned long long)(uintptr_t)gptr;
  v4u g0;
  g0[0] = 1u;                                   // count=1, user descriptor
  g0[1] = lds_addr;                             // LDS byte address
  g0[2] = (unsigned)ga;                         // global_addr[31:0]
  g0[3] = (unsigned)((ga >> 32) & 0x01FFFFFFull) | (2u << 30); // addr[56:32], type=2
  unsigned long long td0 = row_stride_elems;    // tensor_dim0 (full row len)
  unsigned long long td1 = 1ull << 30;          // tensor_dim1 (large, no OOB)
  unsigned long long st0 = row_stride_elems;    // tensor_dim0_stride
  v8i g1;
  g1[0] = (int)(1u << 16);                                          // data_size=2B
  g1[1] = (int)(((unsigned)(td0 & 0xFFFFull)) << 16);               // dim0 lo
  g1[2] = (int)(((td0 >> 16) & 0xFFFFull) | ((td1 & 0xFFFFull) << 16));
  g1[3] = (int)(((td1 >> 16) & 0xFFFFull) | ((unsigned long long)tile_w << 16));
  g1[4] = (int)(tile_h & 0xFFFFu);                                  // tile_dim1
  g1[5] = (int)(st0 & 0xFFFFFFFFull);
  g1[6] = (int)((st0 >> 32) & 0xFFFFull);
  g1[7] = 0;
  v4i z4 = {0, 0, 0, 0};                        // groups 2/3 unused (2D tensor)
  v8i z8 = {0, 0, 0, 0, 0, 0, 0, 0};
  __builtin_amdgcn_tensor_load_to_lds(g0, g1, z4, z4, z8, 0);
}

// ---- fp32 -> fp16 weight conversion ---------------------------------------
__global__ void f32_to_f16_kernel(const float* __restrict__ src,
                                  _Float16* __restrict__ dst, int n) {
  int i = blockIdx.x * 256 + threadIdx.x;
  if (i < n) dst[i] = (_Float16)src[i];
}

// ---- x = x + pos_emb ------------------------------------------------------
__global__ void addpos_kernel(const float* __restrict__ x,
                              const float* __restrict__ pos,
                              float* __restrict__ out, int total, int PD) {
  int i = blockIdx.x * 256 + threadIdx.x;
  if (i < total) out[i] = x[i] + pos[i % PD];
}

// ---- rmsnorm over last dim (D=1024): f16 out, optional f32 out ------------
__global__ __launch_bounds__(256)
void rmsnorm_kernel(const float* __restrict__ x, _Float16* __restrict__ xn,
                    float* __restrict__ xn32, int D) {
  const int row = blockIdx.x;
  const int t = threadIdx.x;
  const float* xr = x + (size_t)row * D;
  float4 v = ((const float4*)xr)[t];
  float ss = v.x * v.x + v.y * v.y + v.z * v.z + v.w * v.w;
#pragma unroll
  for (int d = 16; d > 0; d >>= 1) ss += __shfl_down(ss, d, 32);
  __shared__ float part[8];
  __shared__ float rs;
  if ((t & 31) == 0) part[t >> 5] = ss;
  __syncthreads();
  if (t == 0) {
    float tot = 0.f;
#pragma unroll
    for (int i = 0; i < 8; ++i) tot += part[i];
    rs = rsqrtf(tot / (float)D + 1.1920929e-07f);
  }
  __syncthreads();
  const float r = rs;
  _Float16* xo = xn + (size_t)row * D;
  xo[4 * t + 0] = (_Float16)(v.x * r);
  xo[4 * t + 1] = (_Float16)(v.y * r);
  xo[4 * t + 2] = (_Float16)(v.z * r);
  xo[4 * t + 3] = (_Float16)(v.w * r);
  if (xn32) {
    float4 o; o.x = v.x * r; o.y = v.y * r; o.z = v.z * r; o.w = v.w * r;
    ((float4*)(xn32 + (size_t)row * D))[t] = o;
  }
}

// ---- WMMA GEMM: C[M,N] = A[M,K] @ W[N,K]^T, fused epilogues ---------------
// MODE 0: store f16            outh[m,n]  = acc
// MODE 1: silu-mul             outh[m,n]  = silu(gbuf[m,n]) * acc
// MODE 2: ls residual          xio[m,n]  += ls[n] * acc
// MODE 3: sigmoid gate (final) outf[m,n]  = xf[m,n] * sigmoid(acc)
// Block: 256 threads = 8 waves; 128x64 output tile; K stepped by 64.
// Double-buffered pipeline: wave 0 TDM-loads tile kt+1 into LDS buffer
// (kt+1)&1 while all 8 waves run WMMA on tile kt from buffer kt&1; one
// s_wait_tensorcnt(0) + one workgroup barrier per K tile.
template <int MODE>
__global__ __launch_bounds__(256)
void gemm_wmma_kernel(const _Float16* __restrict__ A,
                      const _Float16* __restrict__ W,
                      int M, int N, int K,
                      _Float16* __restrict__ outh,
                      const _Float16* __restrict__ gbuf,
                      const float* __restrict__ ls,
                      float* __restrict__ xio,
                      const float* __restrict__ xf,
                      float* __restrict__ outf) {
  __shared__ __align__(128) _Float16 sA[2][128 * 64];  // 2 x 16 KB
  __shared__ __align__(128) _Float16 sB[2][64 * 64];   // 2 x 8 KB
  const int wid  = threadIdx.x >> 5;
  const int lane = threadIdx.x & 31;
  const int half = lane >> 4;
  const int lm   = lane & 15;
  const int n0 = blockIdx.x * 64;
  const int m0 = blockIdx.y * 128;

  const v8f vzero = {0.f, 0.f, 0.f, 0.f, 0.f, 0.f, 0.f, 0.f};
  v8f acc[4];
#pragma unroll
  for (int c = 0; c < 4; ++c) acc[c] = vzero;

  const _Float16* Abase = A + (size_t)m0 * K;
  const _Float16* Wbase = W + (size_t)n0 * K;
  const int nk = K >> 6;  // number of 64-wide K tiles

  if (wid == 0) {  // prologue: stage tile 0
    tdm_load_2d((unsigned)(size_t)(void*)sA[0], Abase, 64u, 128u,
                (unsigned long long)K);
    tdm_load_2d((unsigned)(size_t)(void*)sB[0], Wbase, 64u, 64u,
                (unsigned long long)K);
  }

  for (int kt = 0; kt < nk; ++kt) {
    if (wid == 0) {
      // All previously issued TDM ops complete in order; tensorcnt==0 means
      // tile kt has landed in LDS (tile kt+1 not yet issued).
      __builtin_amdgcn_s_wait_tensorcnt(0);
    }
    __syncthreads();  // tile kt visible to all waves; buffer (kt+1)&1 is free

    if (wid == 0 && kt + 1 < nk) {
      const int k0 = (kt + 1) << 6;  // DMA next tile while this one computes
      tdm_load_2d((unsigned)(size_t)(void*)sA[(kt + 1) & 1], Abase + k0, 64u,
                  128u, (unsigned long long)K);
      tdm_load_2d((unsigned)(size_t)(void*)sB[(kt + 1) & 1], Wbase + k0, 64u,
                  64u, (unsigned long long)K);
      if (kt + 2 < nk) {  // pull tile kt+2 toward L2 (global_prefetch)
        __builtin_prefetch(Abase + ((kt + 2) << 6), 0, 1);
        __builtin_prefetch(Wbase + ((kt + 2) << 6), 0, 1);
      }
    }

    const _Float16* a0 = sA[kt & 1];
    const _Float16* b0 = sB[kt & 1];
#pragma unroll
    for (int ks = 0; ks < 2; ++ks) {  // two 32-wide K sub-steps
      // A fragment: 16x32 in this code's contiguous k order — lanes 0..15 hold K=0..15,
      // lanes 16..31 hold K=16..31 for row M = lane%16.
      const _Float16* arow = a0 + (wid * 16 + lm) * 64 + ks * 32;
      v8h alo = *(const v8h*)(arow + half * 16);
      v8h ahi = *(const v8h*)(arow + half * 16 + 8);
      v16h afrag = __builtin_shufflevector(alo, ahi, 0, 1, 2, 3, 4, 5, 6, 7, 8,
                                           9, 10, 11, 12, 13, 14, 15);
#pragma unroll
      for (int c = 0; c < 4; ++c) {
        // B fragment: 32x16 — lanes 0..15 K=0..15, lanes 16..31 K=16..31 for
        // column N = lane%16; W is [N,K] row-major so rows are contiguous.
        v16h bfrag = *(const v16h*)(b0 + (c * 16 + lm) * 64 + ks * 32 +
                                    half * 16);
        acc[c] = __builtin_amdgcn_wmma_f32_16x16x32_f16(
            false, afrag, false, bfrag, (short)0, acc[c], false, false);
      }
    }
    // No trailing barrier needed: the next iteration's top barrier orders all
    // waves' LDS reads (consumed into VGPRs before WMMA) ahead of the next
    // TDM overwrite of this buffer (issued next iteration, after that barrier).
  }

  // Epilogue. C layout: VGPR r, lanes 0..15 -> M=r, lanes 16..31 -> M=8+r.
#pragma unroll
  for (int c = 0; c < 4; ++c) {
#pragma unroll
    for (int r = 0; r < 8; ++r) {
      const int m = m0 + wid * 16 + half * 8 + r;
      const int n = n0 + c * 16 + lm;
      const size_t idx = (size_t)m * N + n;
      const float v = acc[c][r];
      if (MODE == 0) {
        outh[idx] = (_Float16)v;
      } else if (MODE == 1) {
        const float g = (float)gbuf[idx];
        const float sg = g / (1.f + __expf(-g));
        outh[idx] = (_Float16)(sg * v);
      } else if (MODE == 2) {
        xio[idx] += ls[n] * v;
      } else {
        outf[idx] = xf[idx] * (1.f / (1.f + __expf(-v)));
      }
    }
  }
}

// ---- Attention core: one wave per (batch, head); P=16, Dh=256 -------------
// scores = q @ k^T via 8x WMMA f16 (K=256), causal softmax via lane shuffles
// in the documented C-matrix layout, then P @ V in fp32 VALU (tiny).
__global__ __launch_bounds__(32)
void attn_core_kernel(const _Float16* __restrict__ q,
                      const _Float16* __restrict__ k,
                      const _Float16* __restrict__ v,
                      _Float16* __restrict__ att) {
  const int bh = blockIdx.x;
  const int b = bh >> 2, h = bh & 3;
  const int D = 1024, Dh = 256, P = 16;
  const size_t base = (size_t)b * P * D + (size_t)h * Dh;
  const _Float16* qb = q + base;
  const _Float16* kb = k + base;
  const _Float16* vb = v + base;
  _Float16* ob = att + base;
  const int lane = threadIdx.x;
  const int half = lane >> 4;
  const int lm   = lane & 15;

  v8f acc = {0.f, 0.f, 0.f, 0.f, 0.f, 0.f, 0.f, 0.f};
  for (int kk = 0; kk < Dh; kk += 32) {
    const _Float16* ar = qb + (size_t)lm * D + kk;
    v8h alo = *(const v8h*)(ar + half * 16);
    v8h ahi = *(const v8h*)(ar + half * 16 + 8);
    v16h afrag = __builtin_shufflevector(alo, ahi, 0, 1, 2, 3, 4, 5, 6, 7, 8, 9,
                                         10, 11, 12, 13, 14, 15);
    v16h bfrag = *(const v16h*)(kb + (size_t)lm * D + kk + half * 16);
    acc = __builtin_amdgcn_wmma_f32_16x16x32_f16(false, afrag, false, bfrag,
                                                 (short)0, acc, false, false);
  }

  __shared__ float sP[16][16];
  const float scale = 0.0625f;  // Dh^-0.5 = 1/16
#pragma unroll
  for (int r = 0; r < 8; ++r) {
    const int row = half * 8 + r;
    const int col = lm;
    const float sc = acc[r] * scale;
    const bool ok = (col <= row);
    float x = ok ? sc : -3.4e38f;
    float mx = x;
#pragma unroll
    for (int d = 1; d < 16; d <<= 1) mx = fmaxf(mx, __shfl_xor(mx, d, 32));
    float e = ok ? __expf(sc - mx) : 0.f;
    float s = e;
#pragma unroll
    for (int d = 1; d < 16; d <<= 1) s += __shfl_xor(s, d, 32);
    sP[row][col] = e / s;
  }
  __syncthreads();

  // out[i, d] = sum_j P[i,j] * V[j,d]; 32 lanes x 8 columns each.
  const _Float16* vcol = vb + lane * 8;
  for (int i = 0; i < 16; ++i) {
    float o[8] = {0.f, 0.f, 0.f, 0.f, 0.f, 0.f, 0.f, 0.f};
    for (int j = 0; j <= i; ++j) {
      const float pj = sP[i][j];
      const _Float16* vr = vcol + (size_t)j * D;
#pragma unroll
      for (int dd = 0; dd < 8; ++dd) o[dd] += pj * (float)vr[dd];
    }
    _Float16* orow = ob + (size_t)i * D + lane * 8;
#pragma unroll
    for (int dd = 0; dd < 8; ++dd) orow[dd] = (_Float16)o[dd];
  }
}

// ---------------------------------------------------------------------------
extern "C" void kernel_launch(void* const* d_in, const int* in_sizes, int n_in,
                              void* d_out, int out_size, void* d_ws,
                              size_t ws_size, hipStream_t stream) {
  (void)n_in; (void)out_size; (void)ws_size;
  const int B = 1024, P = 16, D = 1024, H = 2048;
  const int M = B * P;  // 16384 tokens

  // ---- workspace carve ----------------------------------------------------
  char* ws = (char*)d_ws;
  size_t off = 0;
  auto alloc = [&](size_t bytes) -> void* {
    void* p = ws + off;
    off = (off + bytes + 255) & ~(size_t)255;
    return p;
  };
  const int widx[18] = {2, 3, 4, 5, 6, 7, 8, 9, 10, 14, 15, 16, 17,
                        19, 20, 21, 22, 24};
  _Float16* wt[32] = {nullptr};
  for (int j = 0; j < 18; ++j) {
    const int i = widx[j];
    wt[i] = (_Float16*)alloc((size_t)in_sizes[i] * sizeof(_Float16));
  }
  float*    bx  = (float*)alloc((size_t)M * D * sizeof(float));      // running x
  _Float16* bxn = (_Float16*)alloc((size_t)M * D * sizeof(_Float16)); // rmsnorm(x)
  _Float16* bg  = (_Float16*)alloc((size_t)M * H * sizeof(_Float16)); // gate proj
  _Float16* bh  = (_Float16*)alloc((size_t)M * H * sizeof(_Float16)); // silu(g)*u
  // attention aliases (swiglu buffers are dead during attention)
  _Float16* bq   = bg;
  _Float16* bk   = bg + (size_t)M * D;
  _Float16* bv   = bh;
  _Float16* batt = bh + (size_t)M * D;

  // ---- weights fp32 -> fp16 ----------------------------------------------
  for (int j = 0; j < 18; ++j) {
    const int i = widx[j];
    const int n = in_sizes[i];
    f32_to_f16_kernel<<<dim3((n + 255) / 256), dim3(256), 0, stream>>>(
        (const float*)d_in[i], wt[i], n);
  }

  // ---- x = x + pos_emb ----------------------------------------------------
  addpos_kernel<<<dim3((M * D + 255) / 256), dim3(256), 0, stream>>>(
      (const float*)d_in[0], (const float*)d_in[1], bx, M * D, P * D);

  const dim3 blk(256);
  const dim3 gH(H / 64, M / 128);  // N=2048 GEMMs
  const dim3 gD(D / 64, M / 128);  // N=1024 GEMMs

  auto swiglu = [&](const _Float16* Wg, const _Float16* Wu, const _Float16* Wd,
                    const float* ls) {
    rmsnorm_kernel<<<dim3(M), blk, 0, stream>>>(bx, bxn, nullptr, D);
    gemm_wmma_kernel<0><<<gH, blk, 0, stream>>>(bxn, Wg, M, H, D, bg, nullptr,
                                                nullptr, nullptr, nullptr,
                                                nullptr);
    gemm_wmma_kernel<1><<<gH, blk, 0, stream>>>(bxn, Wu, M, H, D, bh, bg,
                                                nullptr, nullptr, nullptr,
                                                nullptr);
    gemm_wmma_kernel<2><<<gD, blk, 0, stream>>>(bh, Wd, M, D, H, nullptr,
                                                nullptr, ls, bx, nullptr,
                                                nullptr);
  };
  auto attn = [&](const _Float16* Wq, const _Float16* Wk, const _Float16* Wv,
                  const _Float16* Wo, const float* ls) {
    rmsnorm_kernel<<<dim3(M), blk, 0, stream>>>(bx, bxn, nullptr, D);
    gemm_wmma_kernel<0><<<gD, blk, 0, stream>>>(bxn, Wq, M, D, D, bq, nullptr,
                                                nullptr, nullptr, nullptr,
                                                nullptr);
    gemm_wmma_kernel<0><<<gD, blk, 0, stream>>>(bxn, Wk, M, D, D, bk, nullptr,
                                                nullptr, nullptr, nullptr,
                                                nullptr);
    gemm_wmma_kernel<0><<<gD, blk, 0, stream>>>(bxn, Wv, M, D, D, bv, nullptr,
                                                nullptr, nullptr, nullptr,
                                                nullptr);
    attn_core_kernel<<<dim3(B * 4), dim3(32), 0, stream>>>(bq, bk, bv, batt);
    gemm_wmma_kernel<2><<<gD, blk, 0, stream>>>(batt, Wo, M, D, D, nullptr,
                                                nullptr, ls, bx, nullptr,
                                                nullptr);
  };

  swiglu(wt[2], wt[3], wt[4], (const float*)d_in[11]);             // block a
  swiglu(wt[5], wt[6], wt[7], (const float*)d_in[12]);             // block b
  attn(wt[14], wt[15], wt[16], wt[17], (const float*)d_in[18]);    // attn 1
  swiglu(wt[8], wt[9], wt[10], (const float*)d_in[13]);            // block c
  attn(wt[19], wt[20], wt[21], wt[22], (const float*)d_in[23]);    // attn 2

  // final: x = rmsnorm(x); out = x * sigmoid(x @ Wfg^T)
  rmsnorm_kernel<<<dim3(M), blk, 0, stream>>>(bx, bxn, bx, D);
  gemm_wmma_kernel<3><<<gD, blk, 0, stream>>>(bxn, wt[24], M, D, D, nullptr,
                                              nullptr, nullptr, nullptr, bx,
                                              (float*)d_out);
}
